// CosmosTransformerBlock_88089779240994
// MI455X (gfx1250) — matrix operand order repaired and checked by reading, not compiled, not run on hardware
//
#include <hip/hip_runtime.h>
#include <hip/hip_bf16.h>

typedef __bf16 bf16;
typedef __attribute__((ext_vector_type(16))) __bf16 v16bf;
typedef __attribute__((ext_vector_type(8)))  float  v8f;

#define HID 2048
#define DHEAD 128
#define HEADS 16
#define SEQ 2048
#define SCTX 512
#define FFDIM 8192
#define LORA_R 256
#define E3 (3*HID)

static __device__ inline v8f wmma_bf16(v16bf a, v16bf b, v8f c) {
    // D = A(16x32 bf16) * B(32x16 bf16) + C(16x16 f32)
    return __builtin_amdgcn_wmma_f32_16x16x32_bf16(
        /*neg_a=*/false, a, /*neg_b=*/false, b,
        /*c_mod=*/(short)0, c, /*reuse_a=*/false, /*reuse_b=*/false);
}

// CDNA5 async global->LDS copy (ASYNCcnt-tracked, no VGPR staging).
// lds_off: byte offset into LDS (low 32 bits of generic shared pointer).
static __device__ inline void async_copy_b128(unsigned lds_off, const void* gptr) {
    asm volatile("global_load_async_to_lds_b128 %0, %1, off"
                 :: "v"(lds_off), "v"((unsigned long long)(uintptr_t)gptr)
                 : "memory");
}
static __device__ inline void wait_async0() {
    asm volatile("s_wait_asynccnt 0x0" ::: "memory");
}

// ---------------------------------------------------------------------------
// f32 -> bf16 elementwise (weight staging)
// ---------------------------------------------------------------------------
__global__ void k_f32_to_bf16(const float* __restrict__ x, bf16* __restrict__ y, size_t n) {
    size_t i = (size_t)blockIdx.x * blockDim.x + threadIdx.x;
    size_t stride = (size_t)gridDim.x * blockDim.x;
    for (; i < n; i += stride) y[i] = (bf16)x[i];
}

// ---------------------------------------------------------------------------
// AdaLN LoRA: ld[j] = sum_k silu(emb[k]) * w1[k][j]   (one block per j)
// ---------------------------------------------------------------------------
__global__ __launch_bounds__(256) void k_silu_lora_down(
    const float* __restrict__ emb, const float* __restrict__ w1, float* __restrict__ ld) {
    __shared__ float red[256];
    int j = blockIdx.x, tid = threadIdx.x;
    float acc = 0.f;
    for (int k = tid; k < HID; k += 256) {
        float x = emb[k];
        float s = x / (1.f + __expf(-x));   // silu
        acc += s * w1[(size_t)k * LORA_R + j];
    }
    red[tid] = acc; __syncthreads();
    for (int off = 128; off > 0; off >>= 1) {
        if (tid < off) red[tid] += red[tid + off];
        __syncthreads();
    }
    if (tid == 0) ld[j] = red[0];
}

// e[j] = temb[j] + sum_k ld[k] * w2[k][j],  j in [0, 3*HID)
__global__ void k_lora_up(const float* __restrict__ ld, const float* __restrict__ w2,
                          const float* __restrict__ temb, float* __restrict__ e) {
    int j = blockIdx.x * 256 + threadIdx.x;
    if (j >= E3) return;
    float acc = temb[j];
    for (int k = 0; k < LORA_R; ++k) acc += ld[k] * w2[(size_t)k * E3 + j];
    e[j] = acc;
}

// ---------------------------------------------------------------------------
// LayerNorm (no affine) + AdaLN modulate: Y = LN(x)*(1+scale[c]) + shift[c] -> bf16
// ---------------------------------------------------------------------------
__global__ __launch_bounds__(256) void k_ln_mod(
    const float* __restrict__ X, const float* __restrict__ e, bf16* __restrict__ Y) {
    __shared__ float r1[256], r2[256];
    int row = blockIdx.x, tid = threadIdx.x;
    const float* x = X + (size_t)row * HID;
    float s = 0.f, q = 0.f;
    for (int c = tid; c < HID; c += 256) { float v = x[c]; s += v; q += v * v; }
    r1[tid] = s; r2[tid] = q; __syncthreads();
    for (int off = 128; off > 0; off >>= 1) {
        if (tid < off) { r1[tid] += r1[tid + off]; r2[tid] += r2[tid + off]; }
        __syncthreads();
    }
    float mu  = r1[0] * (1.f / HID);
    float var = r2[0] * (1.f / HID) - mu * mu;
    float rstd = rsqrtf(var + 1e-6f);
    for (int c = tid; c < HID; c += 256) {
        float v = (x[c] - mu) * rstd * (1.f + e[HID + c]) + e[c];
        Y[(size_t)row * HID + c] = (bf16)v;
    }
}

// ---------------------------------------------------------------------------
// Per-head postprocess: optional RMSNorm (over 128) + optional RoPE, then
// write bf16 in head-major [H][S][128] layout. rmsw/cosb may be null.
// ---------------------------------------------------------------------------
__global__ __launch_bounds__(128) void k_head_post(
    const float* __restrict__ X,          // [S][HID]
    const float* __restrict__ rmsw,       // [128] or null
    const float* __restrict__ cosb,       // [S][128] or null
    const float* __restrict__ sinb,       // [S][128] or null
    bf16* __restrict__ Y, int S) {
    __shared__ float red[128];
    __shared__ float yv[128];
    int s = blockIdx.x, h = blockIdx.y, d = threadIdx.x;
    float y = X[(size_t)s * HID + h * DHEAD + d];
    if (rmsw) {
        red[d] = y * y; __syncthreads();
        for (int off = 64; off > 0; off >>= 1) {
            if (d < off) red[d] += red[d + off];
            __syncthreads();
        }
        y = y * rsqrtf(red[0] * (1.f / DHEAD) + 1e-6f) * rmsw[d];
    }
    if (cosb) {
        yv[d] = y; __syncthreads();
        float xr = (d < 64) ? -yv[d + 64] : yv[d - 64];
        y = y * cosb[(size_t)s * DHEAD + d] + xr * sinb[(size_t)s * DHEAD + d];
    }
    Y[((size_t)h * S + s) * DHEAD + d] = (bf16)y;
}

// ---------------------------------------------------------------------------
// GELU exact -> bf16
// ---------------------------------------------------------------------------
__global__ void k_gelu(const float* __restrict__ x, bf16* __restrict__ y, size_t n) {
    size_t i = (size_t)blockIdx.x * blockDim.x + threadIdx.x;
    size_t stride = (size_t)gridDim.x * blockDim.x;
    for (; i < n; i += stride) {
        float v = x[i];
        y[i] = (bf16)(0.5f * v * (1.f + erff(v * 0.70710678118654752f)));
    }
}

// ---------------------------------------------------------------------------
// bf16 WMMA GEMM: C[M,N] = A[M,K] @ B[K,N], A/B bf16 row-major, C f32.
// Block tile 128x128, 8 waves (2x4), each wave 64x32 = 4x2 WMMA tiles.
// A tile staged via CDNA5 async global->LDS B128; B tile staged transposed
// manually (async can't transpose) so B fragments read contiguous b128.
// MODE 0: C = acc.  MODE 1: C = resid + gate[col] * acc  (fused residual).
// ---------------------------------------------------------------------------
template<int MODE>
__global__ __launch_bounds__(256) void k_gemm_bf16(
    const bf16* __restrict__ A, const bf16* __restrict__ B, float* __restrict__ C,
    int M, int N, int K, const float* __restrict__ resid, const float* __restrict__ gate) {
    __shared__ bf16 As[128 * 32];   // [m][k]
    __shared__ bf16 Bs[128 * 32];   // [n][k] (transposed stage)
    const int tid = threadIdx.x;
    const int lane = tid & 31;
    const int w  = tid >> 5;
    const int wm = w >> 2;           // 0..1
    const int wn = w & 3;            // 0..3
    const int m0 = blockIdx.y * 128;
    const int n0 = blockIdx.x * 128;
    const int lm = lane & 15;
    const int lh = lane >> 4;
    const unsigned as_base = (unsigned)(uintptr_t)(&As[0]);

    v8f acc[4][2];
    v8f z = {0.f,0.f,0.f,0.f,0.f,0.f,0.f,0.f};
    for (int i = 0; i < 4; ++i) for (int j = 0; j < 2; ++j) acc[i][j] = z;

    for (int k0 = 0; k0 < K; k0 += 32) {
        // A tile: 128x32 bf16 = 512 x 16B chunks, async direct-to-LDS
        for (int c = tid; c < 512; c += 256) {
            int m = c >> 2, kb8 = c & 3;   // 4 x 16B per 64B row
            async_copy_b128(as_base + c * 16,
                            (const char*)(A + (size_t)(m0 + m) * K + k0) + kb8 * 16);
        }
        // B tile: transposed stage [n][k]
        for (int idx = tid; idx < 128 * 32; idx += 256) {
            int kk = idx >> 7, n = idx & 127;          // coalesced global row read
            Bs[n * 32 + kk] = B[(size_t)(k0 + kk) * N + n0 + n];
        }
        // prefetch next K tile while this one is consumed
        if (k0 + 32 < K) {
            __builtin_prefetch(A + (size_t)(m0 + (tid >> 1)) * K + k0 + 32, 0, 3);
            __builtin_prefetch(B + (size_t)(k0 + 32 + (tid >> 4)) * N + n0 + (tid & 15) * 8, 0, 3);
        }
        wait_async0();
        __syncthreads();

        v16bf af[4], bfr[2];
        for (int mt = 0; mt < 4; ++mt) {
            const bf16* p = &As[(wm * 64 + mt * 16 + lm) * 32 + lh * 16];
            for (int t = 0; t < 16; ++t) af[mt][t] = p[t];
        }
        for (int nt = 0; nt < 2; ++nt) {
            const bf16* p = &Bs[(wn * 32 + nt * 16 + lm) * 32 + lh * 16];
            for (int t = 0; t < 16; ++t) bfr[nt][t] = p[t];
        }
        for (int mt = 0; mt < 4; ++mt)
            for (int nt = 0; nt < 2; ++nt)
                acc[mt][nt] = wmma_bf16(af[mt], bfr[nt], acc[mt][nt]);
        __syncthreads();
    }

    for (int mt = 0; mt < 4; ++mt) {
        for (int nt = 0; nt < 2; ++nt) {
            int col = n0 + wn * 32 + nt * 16 + lm;
            for (int r = 0; r < 8; ++r) {
                int row = m0 + wm * 64 + mt * 16 + lh * 8 + r;
                size_t o = (size_t)row * N + col;
                float v = acc[mt][nt][r];
                if (MODE == 1) v = resid[o] + gate[col] * v;
                C[o] = v;
            }
        }
    }
}

// ---------------------------------------------------------------------------
// Flash attention (non-causal). Q/K/V bf16 head-major [H][S][128].
// Block: 8 waves = 128 query rows of one head; key blocks of 32 staged in LDS
// via CDNA5 async global->LDS copies (ASYNCcnt).
// Output merged bf16 [Sq][HID].
// ---------------------------------------------------------------------------
__global__ __launch_bounds__(256) void k_flash_attn(
    const bf16* __restrict__ Q, const bf16* __restrict__ Km, const bf16* __restrict__ Vm,
    bf16* __restrict__ O, int Sq, int Skv, float scale) {
    __shared__ bf16 Ks[32 * 128];
    __shared__ bf16 Vs[32 * 128];
    __shared__ bf16 Ps[8][16 * 32];
    const int tid = threadIdx.x, lane = tid & 31, w = tid >> 5;
    const int lm = lane & 15, lh = lane >> 4;
    const int h  = blockIdx.y;
    const int q0 = blockIdx.x * 128 + w * 16;
    const bf16* qh = Q  + (size_t)h * Sq  * DHEAD;
    const bf16* kh = Km + (size_t)h * Skv * DHEAD;
    const bf16* vh = Vm + (size_t)h * Skv * DHEAD;
    const unsigned ks_base = (unsigned)(uintptr_t)(&Ks[0]);
    const unsigned vs_base = (unsigned)(uintptr_t)(&Vs[0]);

    // Q fragments stay resident: 4 k-chunks of 32 along D=128 (A layout)
    v16bf qf[4];
    for (int c = 0; c < 4; ++c) {
        const bf16* p = qh + (size_t)(q0 + lm) * DHEAD + c * 32 + lh * 16;
        for (int t = 0; t < 16; ++t) qf[c][t] = p[t];
    }

    v8f z = {0.f,0.f,0.f,0.f,0.f,0.f,0.f,0.f};
    v8f acc[8];
    float mrow[8], lrow[8];
    for (int t = 0; t < 8; ++t) acc[t] = z;
    for (int r = 0; r < 8; ++r) { mrow[r] = -3.0e38f; lrow[r] = 0.f; }

    for (int kb = 0; kb < Skv; kb += 32) {
        // K/V stage: 32x128 bf16 each = 512 x 16B chunks, async direct-to-LDS
        for (int i = tid; i < 512; i += 256) {
            async_copy_b128(ks_base + i * 16, (const char*)(kh + (size_t)kb * DHEAD) + i * 16);
            async_copy_b128(vs_base + i * 16, (const char*)(vh + (size_t)kb * DHEAD) + i * 16);
        }
        wait_async0();
        __syncthreads();

        // S = Q @ K^T : two 16x16 score tiles (keys 0-15, 16-31)
        v8f s0 = z, s1 = z;
        for (int c = 0; c < 4; ++c) {
            v16bf b0, b1;
            const bf16* p0 = &Ks[(lm     ) * DHEAD + c * 32 + lh * 16];
            const bf16* p1 = &Ks[(16 + lm) * DHEAD + c * 32 + lh * 16];
            for (int t = 0; t < 16; ++t) { b0[t] = p0[t]; b1[t] = p1[t]; }
            s0 = wmma_bf16(qf[c], b0, s0);
            s1 = wmma_bf16(qf[c], b1, s1);
        }

        // online softmax per row (rows live per-register, cols across 16 lanes)
        float p0v[8], p1v[8];
        for (int r = 0; r < 8; ++r) {
            float x0 = s0[r] * scale, x1 = s1[r] * scale;
            float mx = fmaxf(x0, x1);
            for (int d = 1; d < 16; d <<= 1) mx = fmaxf(mx, __shfl_xor(mx, d, 32));
            float mn = fmaxf(mrow[r], mx);
            float al = __expf(mrow[r] - mn);
            float e0 = __expf(x0 - mn), e1 = __expf(x1 - mn);
            float sm = e0 + e1;
            for (int d = 1; d < 16; d <<= 1) sm += __shfl_xor(sm, d, 32);
            lrow[r] = lrow[r] * al + sm;
            mrow[r] = mn;
            p0v[r] = e0; p1v[r] = e1;
            for (int t = 0; t < 8; ++t) acc[t][r] *= al;
        }

        // C-layout P -> LDS -> A-layout fragment
        for (int r = 0; r < 8; ++r) {
            int row = lh * 8 + r;
            Ps[w][row * 32 + lm]      = (bf16)p0v[r];
            Ps[w][row * 32 + 16 + lm] = (bf16)p1v[r];
        }
        __syncthreads();
        v16bf pf;
        {
            const bf16* p = &Ps[w][lm * 32 + lh * 16];
            for (int t = 0; t < 16; ++t) pf[t] = p[t];
        }
        // O += P @ V : 8 d-tiles of 16
        for (int dt = 0; dt < 8; ++dt) {
            v16bf vf;
            const bf16* p = &Vs[(lh * 16) * DHEAD + dt * 16 + lm];
            for (int t = 0; t < 16; ++t) vf[t] = p[t * DHEAD];
            acc[dt] = wmma_bf16(pf, vf, acc[dt]);
        }
        __syncthreads();
    }

    for (int r = 0; r < 8; ++r) {
        float inv = 1.f / lrow[r];
        int row = q0 + lh * 8 + r;
        for (int dt = 0; dt < 8; ++dt) {
            int d = dt * 16 + lm;
            O[(size_t)row * HID + h * DHEAD + d] = (bf16)(acc[dt][r] * inv);
        }
    }
}

// ---------------------------------------------------------------------------
// Host orchestration
// ---------------------------------------------------------------------------
extern "C" void kernel_launch(void* const* d_in, const int* in_sizes, int n_in,
                              void* d_out, int out_size, void* d_ws, size_t ws_size,
                              hipStream_t stream) {
    (void)in_sizes; (void)n_in; (void)out_size; (void)ws_size;
    const float* hidden = (const float*)d_in[0];
    const float* enc    = (const float*)d_in[1];
    const float* emb    = (const float*)d_in[2];
    const float* temb   = (const float*)d_in[3];
    const float* rcos   = (const float*)d_in[4];
    const float* rsin   = (const float*)d_in[5];
    const float* a1w1 = (const float*)d_in[6];  const float* a1w2 = (const float*)d_in[7];
    const float* a2w1 = (const float*)d_in[8];  const float* a2w2 = (const float*)d_in[9];
    const float* a3w1 = (const float*)d_in[10]; const float* a3w2 = (const float*)d_in[11];
    const float* w1q = (const float*)d_in[12]; const float* w1k = (const float*)d_in[13];
    const float* w1v = (const float*)d_in[14]; const float* w1o = (const float*)d_in[15];
    const float* qn1 = (const float*)d_in[16]; const float* kn1 = (const float*)d_in[17];
    const float* w2q = (const float*)d_in[18]; const float* w2k = (const float*)d_in[19];
    const float* w2v = (const float*)d_in[20]; const float* w2o = (const float*)d_in[21];
    const float* qn2 = (const float*)d_in[22]; const float* kn2 = (const float*)d_in[23];
    const float* ffw1 = (const float*)d_in[24]; const float* ffw2 = (const float*)d_in[25];

    char* ws = (char*)d_ws;
    size_t off = 0;
    auto alloc = [&](size_t bytes) -> char* {
        char* p = ws + off; off += (bytes + 255) & ~(size_t)255; return p;
    };
    float* ld   = (float*)alloc(LORA_R * 4);
    float* ev   = (float*)alloc(E3 * 4);
    float* gf   = (float*)alloc((size_t)SEQ * FFDIM * 4);       // f32 GEMM scratch
    float* hbuf = (float*)alloc((size_t)SEQ * HID * 4);         // residual stream
    bf16* wbuf = (bf16*)alloc((size_t)HID * FFDIM * 2);         // bf16 weight stage
    bf16* encb = (bf16*)alloc((size_t)SCTX * 1024 * 2);
    bf16* xnb  = (bf16*)alloc((size_t)SEQ * HID * 2);
    bf16* qb   = (bf16*)alloc((size_t)HEADS * SEQ * DHEAD * 2);
    bf16* kbuf = (bf16*)alloc((size_t)HEADS * SEQ * DHEAD * 2);
    bf16* vbuf = (bf16*)alloc((size_t)HEADS * SEQ * DHEAD * 2);
    bf16* mrg  = (bf16*)alloc((size_t)SEQ * HID * 2);
    bf16* gel  = (bf16*)alloc((size_t)SEQ * FFDIM * 2);

    const float scale = 0.08838834764831845f;  // 1/sqrt(128)
    const float* gate = ev + 2 * HID;

    auto cvt = [&](const float* s, bf16* d, size_t n) {
        k_f32_to_bf16<<<dim3(2048), dim3(256), 0, stream>>>(s, d, n);
    };
    auto adaln = [&](const float* w1, const float* w2) {
        k_silu_lora_down<<<dim3(LORA_R), dim3(256), 0, stream>>>(emb, w1, ld);
        k_lora_up<<<dim3(E3 / 256), dim3(256), 0, stream>>>(ld, w2, temb, ev);
    };

    cvt(enc, encb, (size_t)SCTX * 1024);

    // ---------------- self attention ----------------
    adaln(a1w1, a1w2);
    k_ln_mod<<<dim3(SEQ), dim3(256), 0, stream>>>(hidden, ev, xnb);

    cvt(w1q, wbuf, (size_t)HID * HID);
    k_gemm_bf16<0><<<dim3(HID/128, SEQ/128), dim3(256), 0, stream>>>(xnb, wbuf, gf, SEQ, HID, HID, nullptr, nullptr);
    k_head_post<<<dim3(SEQ, HEADS), dim3(128), 0, stream>>>(gf, qn1, rcos, rsin, qb, SEQ);

    cvt(w1k, wbuf, (size_t)HID * HID);
    k_gemm_bf16<0><<<dim3(HID/128, SEQ/128), dim3(256), 0, stream>>>(xnb, wbuf, gf, SEQ, HID, HID, nullptr, nullptr);
    k_head_post<<<dim3(SEQ, HEADS), dim3(128), 0, stream>>>(gf, kn1, rcos, rsin, kbuf, SEQ);

    cvt(w1v, wbuf, (size_t)HID * HID);
    k_gemm_bf16<0><<<dim3(HID/128, SEQ/128), dim3(256), 0, stream>>>(xnb, wbuf, gf, SEQ, HID, HID, nullptr, nullptr);
    k_head_post<<<dim3(SEQ, HEADS), dim3(128), 0, stream>>>(gf, nullptr, nullptr, nullptr, vbuf, SEQ);

    k_flash_attn<<<dim3(SEQ/128, HEADS), dim3(256), 0, stream>>>(qb, kbuf, vbuf, mrg, SEQ, SEQ, scale);

    cvt(w1o, wbuf, (size_t)HID * HID);
    k_gemm_bf16<1><<<dim3(HID/128, SEQ/128), dim3(256), 0, stream>>>(mrg, wbuf, hbuf, SEQ, HID, HID, hidden, gate);

    // ---------------- cross attention ----------------
    adaln(a2w1, a2w2);
    k_ln_mod<<<dim3(SEQ), dim3(256), 0, stream>>>(hbuf, ev, xnb);

    cvt(w2q, wbuf, (size_t)HID * HID);
    k_gemm_bf16<0><<<dim3(HID/128, SEQ/128), dim3(256), 0, stream>>>(xnb, wbuf, gf, SEQ, HID, HID, nullptr, nullptr);
    k_head_post<<<dim3(SEQ, HEADS), dim3(128), 0, stream>>>(gf, qn2, nullptr, nullptr, qb, SEQ);

    cvt(w2k, wbuf, (size_t)1024 * HID);
    k_gemm_bf16<0><<<dim3(HID/128, SCTX/128), dim3(256), 0, stream>>>(encb, wbuf, gf, SCTX, HID, 1024, nullptr, nullptr);
    k_head_post<<<dim3(SCTX, HEADS), dim3(128), 0, stream>>>(gf, kn2, nullptr, nullptr, kbuf, SCTX);

    cvt(w2v, wbuf, (size_t)1024 * HID);
    k_gemm_bf16<0><<<dim3(HID/128, SCTX/128), dim3(256), 0, stream>>>(encb, wbuf, gf, SCTX, HID, 1024, nullptr, nullptr);
    k_head_post<<<dim3(SCTX, HEADS), dim3(128), 0, stream>>>(gf, nullptr, nullptr, nullptr, vbuf, SCTX);

    k_flash_attn<<<dim3(SEQ/128, HEADS), dim3(256), 0, stream>>>(qb, kbuf, vbuf, mrg, SEQ, SCTX, scale);

    cvt(w2o, wbuf, (size_t)HID * HID);
    k_gemm_bf16<1><<<dim3(HID/128, SEQ/128), dim3(256), 0, stream>>>(mrg, wbuf, hbuf, SEQ, HID, HID, hbuf, gate);

    // ---------------- feedforward ----------------
    adaln(a3w1, a3w2);
    k_ln_mod<<<dim3(SEQ), dim3(256), 0, stream>>>(hbuf, ev, xnb);

    cvt(ffw1, wbuf, (size_t)HID * FFDIM);
    k_gemm_bf16<0><<<dim3(FFDIM/128, SEQ/128), dim3(256), 0, stream>>>(xnb, wbuf, gf, SEQ, FFDIM, HID, nullptr, nullptr);
    k_gelu<<<dim3(4096), dim3(256), 0, stream>>>(gf, gel, (size_t)SEQ * FFDIM);

    cvt(ffw2, wbuf, (size_t)FFDIM * HID);
    k_gemm_bf16<1><<<dim3(HID/128, SEQ/128), dim3(256), 0, stream>>>(gel, wbuf, (float*)d_out, SEQ, HID, FFDIM, hbuf, gate);
}
